// GNNEncoder_10299331576571
// MI455X (gfx1250) — matrix operand order repaired and checked by reading, not compiled, not run on hardware
//
#include <hip/hip_runtime.h>
#include <hip/hip_bf16.h>

#define NN 50000
#define NE 800000
#define DIN 96
#define DHID 256
#define DOUT 128

typedef __bf16 v16bf __attribute__((ext_vector_type(16)));
typedef float  v8f   __attribute__((ext_vector_type(8)));

__device__ __forceinline__ unsigned pack_bf16x2(float lo, float hi) {
    unsigned a = __float_as_uint(lo), b = __float_as_uint(hi);
    a += 0x7fffu + ((a >> 16) & 1u);            // RNE
    b += 0x7fffu + ((b >> 16) & 1u);
    // result = {b[31:16], a[31:16]} : bytes = b3,b2,a3,a2 of (hi:lo)
    return __builtin_amdgcn_perm(b, a, 0x07060302u);
}

__device__ __forceinline__ __bf16 to_bf16(float f) {
    unsigned u = __float_as_uint(f);
    u += 0x7fffu + ((u >> 16) & 1u);
    union { unsigned short s; __bf16 b; } out;
    out.s = (unsigned short)(u >> 16);
    return out.b;
}

__global__ void zero_kernel(float* __restrict__ p, long n) {
    long i = (long)blockIdx.x * blockDim.x + threadIdx.x;
    long stride = (long)gridDim.x * blockDim.x;
    for (; i < n; i += stride) p[i] = 0.0f;
}

__global__ void degree_kernel(const int* __restrict__ dst, float* __restrict__ deg, int ne) {
    int i = blockIdx.x * blockDim.x + threadIdx.x;
    int stride = gridDim.x * blockDim.x;
    for (; i < ne; i += stride) atomicAdd(&deg[dst[i]], 1.0f);
}

__global__ void recip_kernel(float* __restrict__ deg, int n) {
    int i = blockIdx.x * blockDim.x + threadIdx.x;
    if (i < n) deg[i] = 1.0f / fmaxf(deg[i], 1.0f);
}

// Concat + transpose weights to bf16: Wt[n][k] for k in [0,2K): Wl then Wr.
template<int K, int NOUT>
__global__ void pack_weights_kernel(const float* __restrict__ Wl,
                                    const float* __restrict__ Wr,
                                    __bf16* __restrict__ Wt) {
    const int Ktot = 2 * K;
    int idx = blockIdx.x * blockDim.x + threadIdx.x;
    if (idx < NOUT * Ktot) {
        int n = idx / Ktot, k = idx % Ktot;
        float v = (k < K) ? Wl[(long)k * NOUT + n] : Wr[(long)(k - K) * NOUT + n];
        Wt[idx] = to_bf16(v);
    }
}

// Gather feat[src[e]] and atomically accumulate into agg[dst[e]], float4 chunks.
template<int D>
__global__ void scatter_kernel(const float* __restrict__ feat,
                               const int* __restrict__ src,
                               const int* __restrict__ dst,
                               float* __restrict__ agg, int ne) {
    const int chunks = D / 4;
    long total = (long)ne * chunks;
    long i = (long)blockIdx.x * blockDim.x + threadIdx.x;
    long stride = (long)gridDim.x * blockDim.x;
    for (; i < total; i += stride) {
        int e = (int)(i / chunks);
        int c = (int)(i % chunks);
        int s = src[e], d = dst[e];
        float4 v = reinterpret_cast<const float4*>(feat + (long)s * D)[c];
        float* o = agg + (long)d * D + c * 4;
        atomicAdd(o + 0, v.x);
        atomicAdd(o + 1, v.y);
        atomicAdd(o + 2, v.z);
        atomicAdd(o + 3, v.w);
    }
}

// out = act( (A_agg * deg_inv) @ Wl + A_self @ Wr + bias )
// LDS-free WMMA GEMM. Block = 8 waves; block tile 128(M) x 64(N).
// Each wave owns rows [wave*16, wave*16+16) x all 64 cols -> 4 accumulators,
// 4 WMMAs per 32-wide K step reusing one A fragment. Branch-free (clamped) loads.
template<int K, int NOUT, bool RELU>
__global__ __launch_bounds__(256)
void sage_gemm_kernel(const float* __restrict__ A_agg,
                      const float* __restrict__ deg_inv,
                      const float* __restrict__ A_self,
                      const __bf16* __restrict__ Wt,   // [NOUT][2K] bf16
                      const float* __restrict__ bias,
                      float* __restrict__ out, int nrows) {
    const int Ktot   = 2 * K;
    const int tid    = threadIdx.x;
    const int wave   = tid >> 5;
    const int lane   = tid & 31;
    const int lane15 = lane & 15;
    const int kh     = lane >> 4;                 // K-half select

    const int m_base  = blockIdx.x * 128 + wave * 16;
    const int n_block = blockIdx.y * 64;

    // A fragment row for this lane (clamped: overflow rows masked at store)
    const int  row_g = m_base + lane15;
    const long row_c = (long)min(row_g, nrows - 1);
    const float dinv = deg_inv[row_c];

    v8f acc[4];
    #pragma unroll
    for (int nt = 0; nt < 4; ++nt) acc[nt] = (v8f){0.f,0.f,0.f,0.f,0.f,0.f,0.f,0.f};

    #pragma unroll
    for (int phase = 0; phase < 2; ++phase) {
        const float* __restrict__ Ap = phase ? A_self : A_agg;
        const float* arow = Ap + row_c * K;
        for (int k0 = 0; k0 < K; k0 += 32) {
            // ---- A fragment: 16 f32 per lane, two 8-float contiguous runs ----
            const float4* ap = (const float4*)(arow + k0 + kh * 8);
            float4 f0 = ap[0];
            float4 f1 = ap[1];
            float4 f2 = ap[4];   // +16 floats
            float4 f3 = ap[5];
            if (phase == 0) {
                f0.x *= dinv; f0.y *= dinv; f0.z *= dinv; f0.w *= dinv;
                f1.x *= dinv; f1.y *= dinv; f1.z *= dinv; f1.w *= dinv;
                f2.x *= dinv; f2.y *= dinv; f2.z *= dinv; f2.w *= dinv;
                f3.x *= dinv; f3.y *= dinv; f3.z *= dinv; f3.w *= dinv;
            }
            union { unsigned u[8]; v16bf v; } a;
            a.u[0] = pack_bf16x2(f0.x, f0.y);
            a.u[1] = pack_bf16x2(f0.z, f0.w);
            a.u[2] = pack_bf16x2(f1.x, f1.y);
            a.u[3] = pack_bf16x2(f1.z, f1.w);
            a.u[4] = pack_bf16x2(f2.x, f2.y);
            a.u[5] = pack_bf16x2(f2.z, f2.w);
            a.u[6] = pack_bf16x2(f3.x, f3.y);
            a.u[7] = pack_bf16x2(f3.z, f3.w);

            // ---- 4 column sub-tiles: B = 16 bf16 per lane, two 8-bf16 runs ----
            #pragma unroll
            for (int nt = 0; nt < 4; ++nt) {
                const int col = n_block + nt * 16 + lane15;
                const uint4* bp = (const uint4*)(Wt + (long)col * Ktot
                                                 + phase * K + k0 + kh * 8);
                union { uint4 q[2]; v16bf v; } b;
                b.q[0] = bp[0];
                b.q[1] = bp[2];
                acc[nt] = __builtin_amdgcn_wmma_f32_16x16x32_bf16(
                              false, a.v, false, b.v, (short)0, acc[nt], false, false);
            }
        }
    }

    // Epilogue: D VGPR j -> row (j + 8*kh); col = lane15. Bias (+ReLU) fused.
    #pragma unroll
    for (int nt = 0; nt < 4; ++nt) {
        const int col = n_block + nt * 16 + lane15;
        const float bv = bias[col];
        #pragma unroll
        for (int j = 0; j < 8; ++j) {
            int row = m_base + j + 8 * kh;
            if (row < nrows) {
                float v = acc[nt][j] + bv;
                if (RELU) v = fmaxf(v, 0.0f);
                out[(long)row * NOUT + col] = v;
            }
        }
    }
}

extern "C" void kernel_launch(void* const* d_in, const int* in_sizes, int n_in,
                              void* d_out, int out_size, void* d_ws, size_t ws_size,
                              hipStream_t stream) {
    (void)in_sizes; (void)n_in; (void)out_size; (void)ws_size;

    const float* x   = (const float*)d_in[0];
    const float* W1l = (const float*)d_in[1];
    const float* b1  = (const float*)d_in[2];
    const float* W1r = (const float*)d_in[3];
    const float* W2l = (const float*)d_in[4];
    const float* b2  = (const float*)d_in[5];
    const float* W2r = (const float*)d_in[6];
    const int*   ei  = (const int*)d_in[7];
    const int* src = ei;            // edge_index[0, :]
    const int* dst = ei + NE;       // edge_index[1, :]
    float* out = (float*)d_out;

    // Workspace: deg_inv | agg1 | h | agg2 | Wt1(bf16) | Wt2(bf16)
    float* deg  = (float*)d_ws;
    float* agg1 = deg  + 50048;
    float* h    = agg1 + (long)NN * DIN;
    float* agg2 = h    + (long)NN * DHID;
    __bf16* Wt1 = (__bf16*)(agg2 + (long)NN * DHID);      // [256][192]
    __bf16* Wt2 = Wt1 + (long)DHID * (2 * DIN);           // [128][512]

    // Zero accumulation buffers (h is fully overwritten by gemm1)
    zero_kernel<<<2048, 256, 0, stream>>>(deg, 50048L + (long)NN * DIN);
    zero_kernel<<<4096, 256, 0, stream>>>(agg2, (long)NN * DHID);

    // bf16 transposed concat weights (tiny, L2-resident)
    pack_weights_kernel<DIN,  DHID><<<(DHID * 2 * DIN  + 255) / 256, 256, 0, stream>>>(W1l, W1r, Wt1);
    pack_weights_kernel<DHID, DOUT><<<(DOUT * 2 * DHID + 255) / 256, 256, 0, stream>>>(W2l, W2r, Wt2);

    // In-degree -> 1/max(deg,1)
    degree_kernel<<<1024, 256, 0, stream>>>(dst, deg, NE);
    recip_kernel<<<(NN + 255) / 256, 256, 0, stream>>>(deg, NN);

    // Layer 1: scatter-sum x, then h = relu(agg1*dinv @ W1l + b1 + x @ W1r)
    scatter_kernel<DIN><<<8192, 256, 0, stream>>>(x, src, dst, agg1, NE);
    dim3 g1((NN + 127) / 128, DHID / 64);
    sage_gemm_kernel<DIN, DHID, true><<<g1, 256, 0, stream>>>(
        agg1, deg, x, Wt1, b1, h, NN);

    // Layer 2: scatter-sum h, then out = agg2*dinv @ W2l + b2 + h @ W2r
    scatter_kernel<DHID><<<16384, 256, 0, stream>>>(h, src, dst, agg2, NE);
    dim3 g2((NN + 127) / 128, DOUT / 64);
    sage_gemm_kernel<DHID, DOUT, false><<<g2, 256, 0, stream>>>(
        agg2, deg, h, Wt2, b2, out, NN);
}
